// GNNClassifier_37245956391182
// MI455X (gfx1250) — matrix operand order repaired and checked by reading, not compiled, not run on hardware
//
#include <hip/hip_runtime.h>
#include <hip/hip_bf16.h>

typedef __attribute__((ext_vector_type(16))) __bf16 v16bf;
typedef __attribute__((ext_vector_type(8)))  float  v8f;

#define HDIM 128

// ---------------------------------------------------------------- utilities
__global__ void gnn_zero(float* __restrict__ p, int n) {
  int i = blockIdx.x * blockDim.x + threadIdx.x;
  int stride = gridDim.x * blockDim.x;
  for (; i < n; i += stride) p[i] = 0.0f;
}

// deg[dst] += 1 per (non-self-loop) edge
__global__ void gnn_degree(const int* __restrict__ dst, float* __restrict__ deg, int E) {
  int i = blockIdx.x * blockDim.x + threadIdx.x;
  if (i < E) atomicAdd(&deg[dst[i]], 1.0f);
}

// dis = rsqrt(deg + 1)   (+1 = self loop; always > 0)
__global__ void gnn_dis(float* __restrict__ degdis, int N) {
  int i = blockIdx.x * blockDim.x + threadIdx.x;
  if (i < N) degdis[i] = rsqrtf(degdis[i] + 1.0f);
}

// ------------------------------------------------- pack W into WMMA B-fragments
// Fragment order: [ktile][colgroup(8)][lane(32)][elem(16)]
// Element (lane, i) of a 32x16 bf16 B tile holds K = (lane&16)+i, N = lane&15.
// Split precision: hi = bf16(w), lo = bf16(w - hi).
__global__ void gnn_pack_w(const float* __restrict__ W, __bf16* __restrict__ hi,
                           __bf16* __restrict__ lo, int K) {
  int idx = blockIdx.x * blockDim.x + threadIdx.x;
  int total = K * HDIM;            // (K/32)*8*32*16
  if (idx >= total) return;
  int i    = idx & 15;
  int lane = (idx >> 4) & 31;
  int cg   = (idx >> 9) & 7;
  int kt   = idx >> 12;
  int k = kt * 32 + (lane & 16) + i;
  int n = cg * 16 + (lane & 15);
  float f = W[k * HDIM + n];
  __bf16 h = (__bf16)f;
  hi[idx] = h;
  lo[idx] = (__bf16)(f - (float)h);
}

// ------------------------------------------------- GEMM: C[N x 128] = A[N x K] @ W
// Block = 128 threads = 4 waves. Block computes a 32-row x 128-col tile.
// Each wave owns 2 row-groups x 2 col-groups = 4 16x16 f32 accumulators.
// B fragments (hi/lo) are loaded once per col-group per K-step and reused across
// both row-groups: 12 WMMAs per 4 KB of B traffic. Split bf16 precision:
// acc += aHi*bHi + aHi*bLo + aLo*bHi  (f32 accumulate, ~fp32 accuracy).
__global__ __launch_bounds__(128) void gnn_gemm(
    const float* __restrict__ A, const __bf16* __restrict__ Bhi,
    const __bf16* __restrict__ Blo, float* __restrict__ C, int K, int Nrows) {
  __shared__ float lds[32 * 33];            // stride 33 -> conflict-free gather
  const int tid  = threadIdx.x;
  const int lane = tid & 31;
  const int wave = tid >> 5;
  const int rowbase = blockIdx.x << 5;      // 32 rows per block
  const int KT = K >> 5;
  const int m = lane & 15;
  const int khalf = (lane & 16) >> 1;       // 0 or 8 (half the A lane-half K offset)

  v8f acc[2][2] = {};                       // [rowgroup][colgroup]

  for (int kt = 0; kt < KT; ++kt) {
    __syncthreads();
#pragma unroll
    for (int ld = 0; ld < 2; ++ld) {        // stage 32x32 f32 A tile, float4 x2/thread
      int idx = tid + (ld << 7);
      int r  = idx >> 3;
      int c4 = (idx & 7) << 2;
      int row = rowbase + r;
      if (row > Nrows - 1) row = Nrows - 1; // clamp (safe duplicate read)
      const float4 v = *reinterpret_cast<const float4*>(
          A + (size_t)row * K + (kt << 5) + c4);
      lds[r * 33 + c4 + 0] = v.x;
      lds[r * 33 + c4 + 1] = v.y;
      lds[r * 33 + c4 + 2] = v.z;
      lds[r * 33 + c4 + 3] = v.w;
    }
    __syncthreads();

    // Build A fragments (16x32 bf16 layout): element i -> K = 2*khalf+i
    v16bf aHi[2], aLo[2];
#pragma unroll
    for (int rg = 0; rg < 2; ++rg) {
#pragma unroll
      for (int i = 0; i < 16; ++i) {
        int k = (khalf << 1) + i;
        float f = lds[(rg * 16 + m) * 33 + k];
        __bf16 h = (__bf16)f;
        aHi[rg][i] = h;
        aLo[rg][i] = (__bf16)(f - (float)h);
      }
    }

#pragma unroll
    for (int cgl = 0; cgl < 2; ++cgl) {
      const int cg = wave * 2 + cgl;
      const size_t fb = ((size_t)(kt * 8 + cg) * 32 + lane) * 16;
      const v16bf bHi = *reinterpret_cast<const v16bf*>(Bhi + fb);
      const v16bf bLo = *reinterpret_cast<const v16bf*>(Blo + fb);
#pragma unroll
      for (int rg = 0; rg < 2; ++rg) {
        acc[rg][cgl] = __builtin_amdgcn_wmma_f32_16x16x32_bf16(
            false, aHi[rg], false, bHi, (short)0, acc[rg][cgl], false, false);
        acc[rg][cgl] = __builtin_amdgcn_wmma_f32_16x16x32_bf16(
            false, aHi[rg], false, bLo, (short)0, acc[rg][cgl], false, false);
        acc[rg][cgl] = __builtin_amdgcn_wmma_f32_16x16x32_bf16(
            false, aLo[rg], false, bHi, (short)0, acc[rg][cgl], false, false);
      }
    }
  }

  // Store: C/D layout: VGPR r -> M = r + 8*(lane>=16), N = lane&15.
  // Bound check hoisted: full tiles (the common case) take the branch-free path.
  const bool full = (rowbase + 32) <= Nrows;
#pragma unroll
  for (int rg = 0; rg < 2; ++rg) {
#pragma unroll
    for (int cgl = 0; cgl < 2; ++cgl) {
      const int col   = (wave * 2 + cgl) * 16 + (lane & 15);
      const int rbase = rowbase + rg * 16 + ((lane & 16) >> 1);
      if (full) {
#pragma unroll
        for (int r = 0; r < 8; ++r)
          C[(size_t)(rbase + r) * HDIM + col] = acc[rg][cgl][r];
      } else {
#pragma unroll
        for (int r = 0; r < 8; ++r)
          if (rbase + r < Nrows)
            C[(size_t)(rbase + r) * HDIM + col] = acc[rg][cgl][r];
      }
    }
  }
}

// --------------------------------------------- self-loop term: out = h * dis^2
__global__ void gnn_selfloop(const float* __restrict__ h, const float* __restrict__ dis,
                             float* __restrict__ out, int total) {
  int i = blockIdx.x * blockDim.x + threadIdx.x;
  int stride = gridDim.x * blockDim.x;
  for (; i < total; i += stride) {
    int node = i >> 7;
    float d = dis[node];
    out[i] = h[i] * d * d;
  }
}

// --------------------------------------------- edge scatter: out[d] += h[s]*dis[s]*dis[d]
// One wave (32 lanes) per edge; each lane handles 4 contiguous columns (float4 load).
// h (51 MB) is L2-resident (192 MB L2), so random row gathers stay on-chip.
__global__ void gnn_scatter(const int* __restrict__ src, const int* __restrict__ dst,
                            const float* __restrict__ dis, const float* __restrict__ h,
                            float* __restrict__ out, int E) {
  const int lane = threadIdx.x & 31;
  const int e = blockIdx.x * (blockDim.x >> 5) + (threadIdx.x >> 5);
  if (e >= E) return;
  const int s = src[e], d = dst[e];
  const float w = dis[s] * dis[d];
  const float4 v = *reinterpret_cast<const float4*>(h + (size_t)s * HDIM + lane * 4);
  float* orow = out + (size_t)d * HDIM + lane * 4;
  atomicAdd(orow + 0, v.x * w);
  atomicAdd(orow + 1, v.y * w);
  atomicAdd(orow + 2, v.z * w);
  atomicAdd(orow + 3, v.w * w);
}

__global__ void gnn_bias_relu(float* __restrict__ out, const float* __restrict__ b, int total) {
  int i = blockIdx.x * blockDim.x + threadIdx.x;
  int stride = gridDim.x * blockDim.x;
  for (; i < total; i += stride) {
    float v = out[i] + b[i & 127];
    out[i] = v > 0.0f ? v : 0.0f;
  }
}

// --------------------------------------------- mean pool (sum + count via atomics)
__global__ void gnn_pool(const float* __restrict__ h, const int* __restrict__ batch,
                         float* __restrict__ sum, float* __restrict__ cnt, int total) {
  int i = blockIdx.x * blockDim.x + threadIdx.x;
  int stride = gridDim.x * blockDim.x;
  for (; i < total; i += stride) {
    int node = i >> 7, c = i & 127;
    int g = batch[node];
    atomicAdd(&sum[g * HDIM + c], h[i]);
    if (c == 0) atomicAdd(&cnt[g], 1.0f);
  }
}

// --------------------------------------------- head: out[G x 2] = emb @ Wh + bh
__global__ void gnn_head(const float* __restrict__ sum, const float* __restrict__ cnt,
                         const float* __restrict__ Wh, const float* __restrict__ bh,
                         float* __restrict__ out, int G) {
  int t = blockIdx.x * blockDim.x + threadIdx.x;
  if (t >= G * 2) return;
  int g = t >> 1, j = t & 1;
  float c = cnt[g];
  if (c < 1.0f) c = 1.0f;
  float inv = 1.0f / c;
  float s = 0.0f;
  for (int k = 0; k < HDIM; ++k) s += sum[g * HDIM + k] * inv * Wh[k * 2 + j];
  out[t] = s + bh[j];
}

// ================================================================ launcher
extern "C" void kernel_launch(void* const* d_in, const int* in_sizes, int n_in,
                              void* d_out, int out_size, void* d_ws, size_t ws_size,
                              hipStream_t stream) {
  (void)n_in; (void)ws_size;
  const float* x   = (const float*)d_in[0];
  const int*  eidx = (const int*)d_in[1];   // JAX x64-off: int64 -> int32
  const int* batch = (const int*)d_in[2];
  const float* W1  = (const float*)d_in[3];
  const float* b1  = (const float*)d_in[4];
  const float* W2  = (const float*)d_in[5];
  const float* b2  = (const float*)d_in[6];
  const float* Wh  = (const float*)d_in[7];
  const float* bh  = (const float*)d_in[8];
  float* out = (float*)d_out;

  const int N  = in_sizes[0] / 768;   // 100000
  const int E  = in_sizes[1] / 2;     // 1600000
  const int G  = out_size / 2;        // 64
  const int K1 = 768;
  const int* src = eidx;
  const int* dst = eidx + E;
  const int total = N * HDIM;

  // ---- workspace carve (all regions fully written before read)
  char* ws = (char*)d_ws;
  size_t off = 0;
  auto alloc = [&](size_t bytes) -> void* {
    void* p = (void*)(ws + off);
    off += (bytes + 255) & ~(size_t)255;
    return p;
  };
  float*  dis  = (float*)alloc((size_t)N * 4);
  float*  bufA = (float*)alloc((size_t)N * HDIM * 4);
  float*  bufB = (float*)alloc((size_t)N * HDIM * 4);
  __bf16* w1hi = (__bf16*)alloc((size_t)K1 * HDIM * 2);
  __bf16* w1lo = (__bf16*)alloc((size_t)K1 * HDIM * 2);
  __bf16* w2hi = (__bf16*)alloc((size_t)HDIM * HDIM * 2);
  __bf16* w2lo = (__bf16*)alloc((size_t)HDIM * HDIM * 2);
  float*  embS = (float*)alloc((size_t)G * HDIM * 4);
  float*  embC = (float*)alloc((size_t)G * 4);

  const int rowtiles = (N + 31) / 32;

  // ---- normalization
  gnn_zero<<<256, 256, 0, stream>>>(dis, N);
  gnn_degree<<<(E + 255) / 256, 256, 0, stream>>>(dst, dis, E);
  gnn_dis<<<(N + 255) / 256, 256, 0, stream>>>(dis, N);

  // ---- pack weights into WMMA fragment order (hi/lo split)
  gnn_pack_w<<<(K1 * HDIM + 255) / 256, 256, 0, stream>>>(W1, w1hi, w1lo, K1);
  gnn_pack_w<<<(HDIM * HDIM + 255) / 256, 256, 0, stream>>>(W2, w2hi, w2lo, HDIM);

  // ---- layer 1: h1 = relu(segsum(norm * (x@W1)) + b1)
  gnn_gemm<<<rowtiles, 128, 0, stream>>>(x, w1hi, w1lo, bufA, K1, N);
  gnn_selfloop<<<2048, 256, 0, stream>>>(bufA, dis, bufB, total);
  gnn_scatter<<<(E + 7) / 8, 256, 0, stream>>>(src, dst, dis, bufA, bufB, E);
  gnn_bias_relu<<<2048, 256, 0, stream>>>(bufB, b1, total);

  // ---- layer 2
  gnn_gemm<<<rowtiles, 128, 0, stream>>>(bufB, w2hi, w2lo, bufA, HDIM, N);
  gnn_selfloop<<<2048, 256, 0, stream>>>(bufA, dis, bufB, total);
  gnn_scatter<<<(E + 7) / 8, 256, 0, stream>>>(src, dst, dis, bufA, bufB, E);
  gnn_bias_relu<<<2048, 256, 0, stream>>>(bufB, b2, total);

  // ---- mean pool + head
  gnn_zero<<<64, 256, 0, stream>>>(embS, G * HDIM);
  gnn_zero<<<1, 256, 0, stream>>>(embC, G);
  gnn_pool<<<2048, 256, 0, stream>>>(bufB, batch, embS, embC, total);
  gnn_head<<<1, 256, 0, stream>>>(embS, embC, Wh, bh, out, G);
}
